// DisentangledSelfAttention_59554016526962
// MI455X (gfx1250) — matrix operand order repaired and checked by reading, compiled, not run on hardware
//
#include <hip/hip_runtime.h>

// ---------------------------------------------------------------------------
// DeBERTa disentangled self-attention for MI455X (gfx1250, wave32, WMMA)
//   B=4, S=1024, H=12, DK=64, HID=768, SPAN=512, P=2*SPAN=1024
// All GEMMs: v_wmma_f32_16x16x32_bf16, 64x16 tile per wave (4x M-blocked).
// Attention GEMMs stage the block-shared 64x64 A tile in LDS via the
// Tensor Data Mover (tensor_load_to_lds + s_wait_tensorcnt) when available.
// ---------------------------------------------------------------------------

#define B_    4
#define S_    1024
#define H_    12
#define DK_   64
#define HID_  768
#define P_    1024               // 2*SPAN
#define BH_   (B_ * H_)          // 48

typedef __attribute__((ext_vector_type(16))) __bf16 v16bf;
typedef __attribute__((ext_vector_type(8)))  float  v8f;

#if defined(__has_builtin)
#if __has_builtin(__builtin_amdgcn_tensor_load_to_lds) && \
    __has_builtin(__builtin_amdgcn_s_wait_tensorcnt)
#define USE_TDM 1
#endif
#endif

#ifdef USE_TDM
typedef __attribute__((ext_vector_type(4))) unsigned int u32x4;
typedef __attribute__((ext_vector_type(8))) int          i32x8;
typedef __attribute__((ext_vector_type(4))) int          i32x4;
#endif

static constexpr float kScale = 0.07216878364870323f; // 1/sqrt(64*3)

// ---- bf16 helpers (storage = unsigned short; hardware convert) ------------
__device__ __forceinline__ unsigned short f2bf(float f) {
    union { __bf16 h; unsigned short s; } u;
    u.h = (__bf16)f;                        // v_cvt_*bf16_f32 on gfx1250
    return u.s;
}
__device__ __forceinline__ float bf2f(unsigned short h) {
    return __uint_as_float(((unsigned int)h) << 16);
}

union V16 {
    unsigned short s[16];
    uint4          u[2];
    v16bf          v;
};

// A fragment (16x32 bf16): lane l -> row = l&15, group g = l>>4.
//   halves 0..7  : K = g*8 + h ; halves 8..15 : K = 16 + g*8 + (h-8)
__device__ __forceinline__ v16bf ldA_bf16(const unsigned short* rowK, int g) {
    V16 t;
    t.u[0] = *(const uint4*)(rowK + g * 8);
    t.u[1] = *(const uint4*)(rowK + 16 + g * 8);
    return t.v;
}
// B fragment (32x16 bf16): lane l -> col = l&15, group g = l>>4.
//   halves 0..7  : K = g*8 + h ; halves 8..15 : K = 16 + g*8 + (h-8)
__device__ __forceinline__ v16bf ldB_bf16(const unsigned short* colK, int g) {
    V16 t;
    t.u[0] = *(const uint4*)(colK + g * 8);
    t.u[1] = *(const uint4*)(colK + 16 + g * 8);
    return t.v;
}

__device__ __forceinline__ v8f wmma_bf16(v16bf a, v16bf b, v8f c) {
    return __builtin_amdgcn_wmma_f32_16x16x32_bf16(
        false, a, false, b, (short)0, c, false, false);
}

// ---------------------------------------------------------------------------
// Kernel 0: elementwise f32 -> bf16 (activations / rel embeddings)
// ---------------------------------------------------------------------------
__global__ void cvt_bf16_kernel(const float* __restrict__ src,
                                unsigned short* __restrict__ dst, int n) {
    int idx = blockIdx.x * 256 + threadIdx.x;
    if (idx < n) dst[idx] = f2bf(src[idx]);
}

// ---------------------------------------------------------------------------
// Kernel 1: transpose + convert five [HID,HID] fp32 weights -> bf16 [N,K]
// ---------------------------------------------------------------------------
__global__ void wt_convert_kernel(const float* W0, const float* W1, const float* W2,
                                  const float* W3, const float* W4,
                                  unsigned short* T0, unsigned short* T1, unsigned short* T2,
                                  unsigned short* T3, unsigned short* T4) {
    const float* Ws[5] = {W0, W1, W2, W3, W4};
    unsigned short* Ts[5] = {T0, T1, T2, T3, T4};
    int m   = blockIdx.y;
    int idx = blockIdx.x * 256 + threadIdx.x;
    int n   = idx / HID_;
    int k   = idx % HID_;
    Ts[m][(size_t)n * HID_ + k] = f2bf(Ws[m][(size_t)k * HID_ + n]);
}

// ---------------------------------------------------------------------------
// Kernel 2: projection GEMM  dst = A[M,768] @ W + bias (bf16 in, bf16 out)
//   Each wave: 64(M) x 16(N); B fragment loaded once per K-step, reused x4.
//   Row decomposition uses S==1024 (shift/mask, no integer division).
//   storeTransposed==0: dst[((b*H+h)*1024 + s)*64 + d]
//   storeTransposed==1: dst[((b*H+h)*64 + d)*1024 + s]   (v for ctx B-operand)
// ---------------------------------------------------------------------------
__global__ void proj_kernel(const unsigned short* __restrict__ A,
                            const unsigned short* __restrict__ WT,
                            const float* __restrict__ bias,
                            unsigned short* __restrict__ dst,
                            int M, int storeTransposed) {
    int t = blockIdx.x * 8 + threadIdx.y;              // wave id
    int waves = (M / 64) * (HID_ / 16);
    if (t >= waves) return;                            // wave-uniform (exact grid)
    int nt = t % (HID_ / 16);
    int mb = t / (HID_ / 16);
    int lane = threadIdx.x;
    int g  = lane >> 4;
    int lm = lane & 15;

    const unsigned short* brow  = WT + (size_t)(nt * 16 + lm) * HID_;
    const unsigned short* arow0 = A  + (size_t)(mb * 64 + lm) * HID_;

    v8f acc[4] = {};
#pragma unroll 2
    for (int kk = 0; kk < HID_; kk += 32) {
        __builtin_prefetch(brow + kk + 128, 0, 1);     // global_prefetch_b8
        v16bf b = ldB_bf16(brow + kk, g);
#pragma unroll
        for (int s = 0; s < 4; ++s) {
            v16bf a = ldA_bf16(arow0 + (size_t)s * 16 * HID_ + kk, g);
            acc[s] = wmma_bf16(a, b, acc[s]);
        }
    }

    int col = nt * 16 + lm;
    float bv = bias[col];
    int h = col >> 6;
    int d = col & 63;
#pragma unroll
    for (int s = 0; s < 4; ++s) {
#pragma unroll
        for (int r = 0; r < 8; ++r) {
            int row  = mb * 64 + s * 16 + g * 8 + r;
            int bidx = row >> 10;                      // row / 1024
            int s2   = row & 1023;                     // row % 1024
            float val = acc[s][r] + bv;
            size_t o = storeTransposed
                ? ((size_t)(bidx * H_ + h) * DK_ + d) * 1024 + s2
                : ((size_t)(bidx * H_ + h) * 1024 + s2) * DK_ + d;
            dst[o] = f2bf(val);
        }
    }
}

// ---------------------------------------------------------------------------
// Kernel 3: attention GEMM over DK:  out[m,n] = scale * A[m,:]·Bm[n,:]
//   Grid (ntg, mb, bh): 8 waves/block share one 64x64 bf16 A tile, staged in
//   LDS once (TDM tensor_load_to_lds when available, manual copy otherwise).
//   Each wave: 64(M) x 16(N), K=64 -> 8 WMMAs, B frags loaded once.
//   mode 0: f32 scores -> outF ; mode 1: bf16 -> outB (c2p/p2c scratch)
// ---------------------------------------------------------------------------
__global__ void attn_gemm_kernel(const unsigned short* __restrict__ Abase,
                                 const unsigned short* __restrict__ Bbase,
                                 int bHeadOnly,
                                 float* __restrict__ outF,
                                 unsigned short* __restrict__ outB,
                                 int mode, float scale) {
    __shared__ unsigned short atile[64 * 64];          // 8 KB, LDS offset 0

    int ntg = blockIdx.x;                              // 0..7
    int mb  = blockIdx.y;                              // 0..15
    int bh  = blockIdx.z;                              // 0..47

    const unsigned short* A  = Abase + ((size_t)bh * S_ + (size_t)mb * 64) * DK_; // tile start
    const unsigned short* Bm = Bbase + (size_t)(bHeadOnly ? (bh % H_) : bh) * (size_t)(1024 * DK_);

#ifdef USE_TDM
    if (threadIdx.y == 0) {                            // one wave issues the TDM op
        unsigned long long ga = (unsigned long long)(const void*)A;
        u32x4 g0;
        g0[0] = 0x1u;                                  // count=1, user mode
        g0[1] = 0u;                                    // lds_addr = 0 (atile)
        g0[2] = (unsigned)(ga & 0xFFFFFFFFu);          // global_addr[31:0]
        g0[3] = (unsigned)((ga >> 32) & 0x01FFFFFFu)   // global_addr[56:32]
              | 0x80000000u;                           // type = 2 ("image")
        i32x8 g1;
        g1[0] = 0x00010000;                            // data_size = 1 (2 bytes)
        g1[1] = (64 << 16);                            // tensor_dim0 = 64
        g1[2] = (64 << 16);                            // tensor_dim1 = 64
        g1[3] = (64 << 16);                            // tile_dim0   = 64
        g1[4] = 64;                                    // tile_dim1   = 64
        g1[5] = 64;                                    // tensor_dim0_stride = 64
        g1[6] = 0;
        g1[7] = 0;
        i32x4 z4 = {0, 0, 0, 0};                       // groups 2/3 unused (2D)
        i32x8 z8 = {0, 0, 0, 0, 0, 0, 0, 0};
        __builtin_amdgcn_tensor_load_to_lds(g0, g1, z4, z4, z8, 0);
        __builtin_amdgcn_s_wait_tensorcnt(0);
    }
#else
    {
        int tid = threadIdx.y * 32 + threadIdx.x;      // 0..255
#pragma unroll
        for (int c = tid; c < 512; c += 256) {         // 512 x 16B chunks
            int row = c >> 2;
            int off = (c & 3) * 8;
            *(uint4*)&atile[row * 64 + off] = *(const uint4*)(A + row * DK_ + off);
        }
    }
#endif
    __syncthreads();

    int lane = threadIdx.x;
    int g  = lane >> 4;
    int lm = lane & 15;
    int nt = ntg * 8 + threadIdx.y;

    const unsigned short* brow = Bm + (size_t)(nt * 16 + lm) * DK_;
    v16bf b0 = ldB_bf16(brow, g);
    v16bf b1 = ldB_bf16(brow + 32, g);

    v8f acc[4] = {};
#pragma unroll
    for (int s = 0; s < 4; ++s) {
        const unsigned short* arow = &atile[(s * 16 + lm) * 64];  // ds_load_b128
        v16bf a0 = ldA_bf16(arow, g);
        v16bf a1 = ldA_bf16(arow + 32, g);
        acc[s] = wmma_bf16(a0, b0, acc[s]);
        acc[s] = wmma_bf16(a1, b1, acc[s]);
    }

#pragma unroll
    for (int s = 0; s < 4; ++s) {
        size_t base = ((size_t)bh * S_ + (size_t)(mb * 64 + s * 16 + g * 8)) * 1024
                    + (size_t)(nt * 16 + lm);
        if (mode == 0) {
#pragma unroll
            for (int r = 0; r < 8; ++r) outF[base + (size_t)r * 1024] = acc[s][r] * scale;
        } else {
#pragma unroll
            for (int r = 0; r < 8; ++r) outB[base + (size_t)r * 1024] = f2bf(acc[s][r] * scale);
        }
    }
}

// ---------------------------------------------------------------------------
// Kernel 4: fused gather + mask + softmax. Writes f32 weights (output) and a
// bf16 copy (ws) consumed by the ctx GEMM.
//   scores[q,k] += c2p[q,d] + p2c[k,d],  d = clip(q-k+512, 0, 1023)
// ---------------------------------------------------------------------------
__global__ void softmax_kernel(float* __restrict__ W,
                               unsigned short* __restrict__ Wbf,
                               const unsigned short* __restrict__ c2p,
                               const unsigned short* __restrict__ p2c,
                               const float* __restrict__ mask) {
    int row = blockIdx.x;                  // bh*S + q
    int bh  = row >> 10;
    int q   = row & 1023;
    int b   = bh / H_;

    float*                srow   = W   + (size_t)row * S_;
    unsigned short*       wrow   = Wbf + (size_t)row * S_;
    const unsigned short* c2prow = c2p + (size_t)row * P_;
    const unsigned short* p2cb   = p2c + (size_t)bh * S_ * P_;
    const float*          mrow   = mask + (size_t)b * S_;

    int tid = threadIdx.x;
    float vals[4];
    float mx = -1e30f;
#pragma unroll
    for (int i = 0; i < 4; ++i) {
        int k = i * 256 + tid;
        int d = q - k + 512;
        d = d < 0 ? 0 : (d > 1023 ? 1023 : d);
        float v = srow[k]
                + bf2f(c2prow[d])
                + bf2f(p2cb[(size_t)k * P_ + d])
                + mrow[k] * (-1e9f);
        vals[i] = v;
        mx = fmaxf(mx, v);
    }

    __shared__ float red[256];
    red[tid] = mx;
    __syncthreads();
#pragma unroll
    for (int s2 = 128; s2 > 0; s2 >>= 1) {
        if (tid < s2) red[tid] = fmaxf(red[tid], red[tid + s2]);
        __syncthreads();
    }
    mx = red[0];
    __syncthreads();

    float sum = 0.f;
#pragma unroll
    for (int i = 0; i < 4; ++i) {
        vals[i] = __expf(vals[i] - mx);
        sum += vals[i];
    }
    red[tid] = sum;
    __syncthreads();
#pragma unroll
    for (int s2 = 128; s2 > 0; s2 >>= 1) {
        if (tid < s2) red[tid] += red[tid + s2];
        __syncthreads();
    }
    float inv = 1.0f / red[0];
#pragma unroll
    for (int i = 0; i < 4; ++i) {
        int k = i * 256 + tid;
        float w = vals[i] * inv;
        srow[k] = w;
        wrow[k] = f2bf(w);
    }
}

// ---------------------------------------------------------------------------
// Kernel 5: ctx = weights[S,S] @ v[S,64] per (b,h).  A = bf16 weights copy,
// B = vT [BH,64,S].  Each wave: 64(M) x 16(N), K-loop of 32.
// ---------------------------------------------------------------------------
__global__ void ctx_kernel(const unsigned short* __restrict__ Wbf,
                           const unsigned short* __restrict__ vT,
                           float* __restrict__ ctx) {
    int t = blockIdx.x * 8 + threadIdx.y;              // BH * 16 * 4 waves
    if (t >= BH_ * 16 * 4) return;
    int nt = t % 4;
    int mb = (t / 4) % 16;
    int bh = t / (4 * 16);

    int lane = threadIdx.x;
    int g  = lane >> 4;
    int lm = lane & 15;
    const unsigned short* arow0 = Wbf + ((size_t)bh * S_ + (size_t)(mb * 64 + lm)) * S_;
    const unsigned short* brow  = vT  + ((size_t)bh * DK_ + (size_t)(nt * 16 + lm)) * S_;

    v8f acc[4] = {};
#pragma unroll 2
    for (int kk = 0; kk < S_; kk += 32) {
        __builtin_prefetch(arow0 + kk + 256, 0, 1);
        v16bf b = ldB_bf16(brow + kk, g);
#pragma unroll
        for (int s = 0; s < 4; ++s) {
            v16bf a = ldA_bf16(arow0 + (size_t)s * 16 * S_ + kk, g);
            acc[s] = wmma_bf16(a, b, acc[s]);
        }
    }

    int b = bh / H_;
    int h = bh % H_;
    int d = nt * 16 + lm;
#pragma unroll
    for (int s = 0; s < 4; ++s) {
#pragma unroll
        for (int r = 0; r < 8; ++r) {
            int qq = mb * 64 + s * 16 + g * 8 + r;
            ctx[((size_t)b * S_ + qq) * HID_ + h * DK_ + d] = acc[s][r];
        }
    }
}

// ---------------------------------------------------------------------------
// Workspace layout (bytes, all 256-aligned)
// ---------------------------------------------------------------------------
static constexpr size_t WT_BYTES   = (size_t)HID_ * HID_ * 2;        // 1,179,648
static constexpr size_t ACT_BYTES  = (size_t)B_ * S_ * HID_ * 2;     // 6,291,456
static constexpr size_t REL_IN_B   = (size_t)P_ * HID_ * 2;          // 1,572,864
static constexpr size_t QH_BYTES   = (size_t)BH_ * S_ * DK_ * 2;     // 6,291,456
static constexpr size_t POS_BYTES  = (size_t)H_ * P_ * DK_ * 2;      // 1,572,864
static constexpr size_t REL_BYTES  = (size_t)BH_ * S_ * P_ * 2;      // 100,663,296

static constexpr size_t OFF_WT   = 0;                                // 5 matrices
static constexpr size_t OFF_QBF  = OFF_WT + 5 * WT_BYTES;
static constexpr size_t OFF_KBF  = OFF_QBF + ACT_BYTES;
static constexpr size_t OFF_VBF  = OFF_KBF + ACT_BYTES;
static constexpr size_t OFF_REBF = OFF_VBF + ACT_BYTES;
static constexpr size_t OFF_QH   = OFF_REBF + REL_IN_B;
static constexpr size_t OFF_KH   = OFF_QH + QH_BYTES;
static constexpr size_t OFF_VT   = OFF_KH + QH_BYTES;
static constexpr size_t OFF_POSK = OFF_VT + QH_BYTES;
static constexpr size_t OFF_POSQ = OFF_POSK + POS_BYTES;
static constexpr size_t OFF_C2P  = OFF_POSQ + POS_BYTES;
static constexpr size_t OFF_P2C  = OFF_C2P + REL_BYTES;
static constexpr size_t OFF_WBF  = OFF_P2C + REL_BYTES;

static constexpr size_t CTX_ELEMS = (size_t)B_ * S_ * HID_;          // 3,145,728

extern "C" void kernel_launch(void* const* d_in, const int* in_sizes, int n_in,
                              void* d_out, int out_size, void* d_ws, size_t ws_size,
                              hipStream_t stream) {
    const float* query = (const float*)d_in[0];
    const float* key   = (const float*)d_in[1];
    const float* value = (const float*)d_in[2];
    const float* rel   = (const float*)d_in[3];
    const float* mask  = (const float*)d_in[4];
    // d_in[5] = relative_pos (int32): recomputed analytically (clip(q-k+512))
    const float* Wq  = (const float*)d_in[6];   const float* bq  = (const float*)d_in[7];
    const float* Wk  = (const float*)d_in[8];   const float* bk  = (const float*)d_in[9];
    const float* Wv  = (const float*)d_in[10];  const float* bv  = (const float*)d_in[11];
    const float* Wpk = (const float*)d_in[12];  const float* bpk = (const float*)d_in[13];
    const float* Wpq = (const float*)d_in[14];  const float* bpq = (const float*)d_in[15];

    float* out = (float*)d_out;
    float* ctx = out;                       // [B,S,768]
    float* wts = out + CTX_ELEMS;           // [B,H,S,S]

    char* ws = (char*)d_ws;
    unsigned short* WqT  = (unsigned short*)(ws + OFF_WT + 0 * WT_BYTES);
    unsigned short* WkT  = (unsigned short*)(ws + OFF_WT + 1 * WT_BYTES);
    unsigned short* WvT  = (unsigned short*)(ws + OFF_WT + 2 * WT_BYTES);
    unsigned short* WpkT = (unsigned short*)(ws + OFF_WT + 3 * WT_BYTES);
    unsigned short* WpqT = (unsigned short*)(ws + OFF_WT + 4 * WT_BYTES);
    unsigned short* qbf  = (unsigned short*)(ws + OFF_QBF);
    unsigned short* kbf  = (unsigned short*)(ws + OFF_KBF);
    unsigned short* vbf  = (unsigned short*)(ws + OFF_VBF);
    unsigned short* rebf = (unsigned short*)(ws + OFF_REBF);
    unsigned short* qh   = (unsigned short*)(ws + OFF_QH);
    unsigned short* kh   = (unsigned short*)(ws + OFF_KH);
    unsigned short* vT   = (unsigned short*)(ws + OFF_VT);
    unsigned short* posk = (unsigned short*)(ws + OFF_POSK);
    unsigned short* posq = (unsigned short*)(ws + OFF_POSQ);
    unsigned short* c2p  = (unsigned short*)(ws + OFF_C2P);
    unsigned short* p2c  = (unsigned short*)(ws + OFF_P2C);
    unsigned short* wbf  = (unsigned short*)(ws + OFF_WBF);

    dim3 blk(32, 8);                                   // 8 wave32 per block

    // 0) activation converts (f32 -> bf16, one pass each)
    const int NACT = B_ * S_ * HID_;                   // 3,145,728
    const int NREL = P_ * HID_;                        // 786,432
    cvt_bf16_kernel<<<NACT / 256, 256, 0, stream>>>(query, qbf, NACT);
    cvt_bf16_kernel<<<NACT / 256, 256, 0, stream>>>(key,   kbf, NACT);
    cvt_bf16_kernel<<<NACT / 256, 256, 0, stream>>>(value, vbf, NACT);
    cvt_bf16_kernel<<<NREL / 256, 256, 0, stream>>>(rel,  rebf, NREL);

    // 1) weight transpose/convert
    wt_convert_kernel<<<dim3((HID_ * HID_) / 256, 5), 256, 0, stream>>>(
        Wq, Wk, Wv, Wpk, Wpq, WqT, WkT, WvT, WpkT, WpqT);

    // 2) projections: waves = (M/64)*48 -> blocks = waves/8
    proj_kernel<<<384, blk, 0, stream>>>(qbf,  WqT,  bq,  qh,   4096, 0);
    proj_kernel<<<384, blk, 0, stream>>>(kbf,  WkT,  bk,  kh,   4096, 0);
    proj_kernel<<<384, blk, 0, stream>>>(vbf,  WvT,  bv,  vT,   4096, 1);
    proj_kernel<<<96,  blk, 0, stream>>>(rebf, WpkT, bpk, posk, 1024, 0);
    proj_kernel<<<96,  blk, 0, stream>>>(rebf, WpqT, bpq, posq, 1024, 0);

    // 3) content scores (f32), c2p / p2c (bf16); LDS-staged A tile per block
    dim3 agrid(8, 16, 48);
    attn_gemm_kernel<<<agrid, blk, 0, stream>>>(qh, kh,   0, wts, nullptr, 0, kScale);
    attn_gemm_kernel<<<agrid, blk, 0, stream>>>(qh, posk, 1, nullptr, c2p, 1, kScale);
    attn_gemm_kernel<<<agrid, blk, 0, stream>>>(kh, posq, 1, nullptr, p2c, 1, kScale);

    // 4) fused gather + mask + softmax (f32 out + bf16 copy)
    softmax_kernel<<<BH_ * S_, 256, 0, stream>>>(wts, wbf, c2p, p2c, mask);

    // 5) ctx = weights @ v : waves = 48*16*4 -> 384 blocks
    ctx_kernel<<<384, blk, 0, stream>>>(wbf, vT, ctx);

    (void)in_sizes; (void)n_in; (void)out_size; (void)ws_size;
}
